// StepWiseMLPAutoEncoder_2826088480934
// MI455X (gfx1250) — matrix operand order repaired and checked by reading, not compiled, not run on hardware
//
#include <hip/hip_runtime.h>
#include <hip/hip_bf16.h>

// ---------------------------------------------------------------------------
// StepWiseMLPAutoEncoder on gfx1250 (MI455X), wave32 + WMMA f16->f32.
//
//   Encoder (parallel over T): X0[16384,2080p] -> 1408p -> 768p -> 64 (tanh)
//   Decoder: ONE persistent kernel, 512 sequential steps with grid-wide
//   barriers between layers (5 resident blocks, L2-resident f16 weights).
//
// WMMA core: wave tile 16(M)x64(N), 4 f32 accumulators; per K-tile =
// 2x b128 A-loads + 8x b128 B-loads (packed, immediate offsets) + 4x
// v_wmma_f32_16x16x32_f16; K-loop unrolled 2x for load/wmma overlap.
// ---------------------------------------------------------------------------

typedef __attribute__((ext_vector_type(16))) _Float16 v16h;
typedef __attribute__((ext_vector_type(8)))  _Float16 v8h;
typedef __attribute__((ext_vector_type(8)))  float    v8f;

namespace sw {
constexpr int B = 32, F = 1024, T = 512, H = 64;
// real dims
constexpr int KE  = 2*F + 1;                 // 2049
constexpr int NE1 = 1387, NE2 = 726, NE3 = H;
constexpr int KD  = F + H + 1;               // 1089
constexpr int ND1 = 1067, ND2 = 1046, ND3 = F;
constexpr int ME  = T * B;                   // 16384 (mult of 32)
// padded dims (K -> mult 32, N -> mult 64)
constexpr int KEp  = 2080;
constexpr int NE1p = 1408, NE2p = 768, NE3p = 64;
constexpr int KDp  = 1120;
constexpr int ND1p = 1088, ND2p = 1088, ND3p = 1024;
constexpr int NBLK = 5;                      // persistent decoder blocks (1088/256 -> 5)
}

static __device__ inline v16h cat8(v8h lo, v8h hi) {
  return __builtin_shufflevector(lo, hi, 0,1,2,3,4,5,6,7,8,9,10,11,12,13,14,15);
}

// --------------------------- wave-level WMMA GEMM ---------------------------
// One wave computes rows [m0, m0+16) x cols [n0, n0+64) of
// C[M, Np] = act(A[., Kp] @ Wp + bias); Wp pre-packed:
//   Wp[((kt*2 + h) * Np + n) * 16 + i] = W[kt*32 + h*16 + i][n]
// ACT: 0=none, 1=relu, 2=tanh.  Caller guarantees n0 + 64 <= Np, m0 + 16 <= M.
template <int ACT>
static __device__ inline void wave_gemm(
    const _Float16* __restrict__ A, const _Float16* __restrict__ Wp,
    const float* __restrict__ bias,
    _Float16* __restrict__ outH, int outH_ld,
    float* __restrict__ outF, long fRowStride, long fColStride,
    int Kp, int Np, int Nreal, int m0, int n0, int lane) {
  const int half  = (lane >> 4) & 1;          // 0: lanes 0-15, 1: lanes 16-31
  const int mrow  = m0 + (lane & 15);
  const int ncol0 = n0 + (lane & 15);

  const _Float16* aPtr = A + (size_t)mrow * (size_t)Kp + (size_t)(half * 16);
  const _Float16* bPtr = Wp + ((size_t)half * (size_t)Np + (size_t)ncol0) * 16;
  const size_t bStep = (size_t)2 * (size_t)Np * 16;

  v8f acc[4] = {v8f{}, v8f{}, v8f{}, v8f{}};
  const int ktiles = Kp >> 5;
#pragma unroll 2
  for (int kt = 0; kt < ktiles; ++kt) {
    v16h a = cat8(*(const v8h*)(aPtr), *(const v8h*)(aPtr + 8));
    v16h bf[4];
#pragma unroll
    for (int j = 0; j < 4; ++j)
      bf[j] = cat8(*(const v8h*)(bPtr + j * 256), *(const v8h*)(bPtr + j * 256 + 8));
#pragma unroll
    for (int j = 0; j < 4; ++j)
      acc[j] = __builtin_amdgcn_wmma_f32_16x16x32_f16(
          false, a, false, bf[j], (short)0, acc[j], false, false);
    aPtr += 32;
    bPtr += bStep;
  }

  const int rbase = m0 + (half ? 8 : 0);
#pragma unroll
  for (int j = 0; j < 4; ++j) {
    const int ncol = ncol0 + j * 16;
    const float bv = (bias && ncol < Nreal) ? bias[ncol] : 0.f;
#pragma unroll
    for (int r = 0; r < 8; ++r) {
      const int row = rbase + r;
      float v = acc[j][r] + bv;
      if (ACT == 1) v = fmaxf(v, 0.f);
      if (ACT == 2) v = tanhf(v);
      outH[(size_t)row * (size_t)outH_ld + ncol] = (_Float16)v;   // padded cols -> 0
      if (outF && ncol < Nreal)
        outF[(size_t)row * (size_t)fRowStride + (size_t)ncol * (size_t)fColStride] = v;
    }
  }
}

// ------------------------- standalone GEMM (encoder) ------------------------
template <int ACT>
__global__ __launch_bounds__(256)
void gemm_f16_wmma(const _Float16* __restrict__ A, const _Float16* __restrict__ Wp,
                   const float* __restrict__ bias,
                   _Float16* __restrict__ outH, int outH_ld,
                   int Kp, int Np, int Nreal) {
  const int lane = threadIdx.x & 31;
  const int wave = threadIdx.x >> 5;
  const int m0   = blockIdx.y * 32  + (wave >> 2) * 16;
  const int n0   = blockIdx.x * 256 + (wave & 3) * 64;
  if (n0 >= Np) return;                       // uniform per wave; Np % 64 == 0
  wave_gemm<ACT>(A, Wp, bias, outH, outH_ld, nullptr, 0, 0,
                 Kp, Np, Nreal, m0, n0, lane);
}

// ------------------------- persistent decoder kernel ------------------------
// Grid-wide barrier: release-fence all stores, then one rep per block bumps an
// L2 counter and spins (s_sleep) until all NBLK blocks arrived.
static __device__ inline void grid_barrier(unsigned* __restrict__ ctr, unsigned target) {
  __threadfence();                                   // release: stores -> device scope
  __syncthreads();
  if (threadIdx.x == 0) {
    atomicAdd(ctr, 1u);
    while (__hip_atomic_load(ctr, __ATOMIC_RELAXED, __HIP_MEMORY_SCOPE_AGENT) < target)
      __builtin_amdgcn_s_sleep(2);
  }
  __syncthreads();
  __threadfence();                                   // acquire: see other blocks' stores
}

__global__ void __launch_bounds__(256)
decoder_persistent(const _Float16* __restrict__ wD0, const float* __restrict__ db0,
                   const _Float16* __restrict__ wD1, const float* __restrict__ db1,
                   const _Float16* __restrict__ wD2, const float* __restrict__ db2,
                   _Float16* __restrict__ DIN, _Float16* __restrict__ dact1,
                   _Float16* __restrict__ dact2, float* __restrict__ y,
                   unsigned* __restrict__ ctr) {
  using namespace sw;
  const int lane = threadIdx.x & 31;
  const int wave = threadIdx.x >> 5;
  const int m0   = (wave >> 2) * 16;                 // M = 32 -> 2 wave rows
  const int n0   = blockIdx.x * 256 + (wave & 3) * 64;
  unsigned barNum = 0;

  for (int t = 0; t < T; ++t) {
    const _Float16* din_t = DIN + (size_t)t * B * KDp;
    _Float16* fb_next     = DIN + (size_t)(t + 1) * B * KDp + H;  // cols 64..1087

    if (n0 < ND1p)
      wave_gemm<1>(din_t, wD0, db0, dact1, ND1p, nullptr, 0, 0,
                   KDp, ND1p, ND1, m0, n0, lane);
    grid_barrier(ctr, ++barNum * NBLK);

    if (n0 < ND2p)
      wave_gemm<1>(dact1, wD1, db1, dact2, ND2p, nullptr, 0, 0,
                   ND1p, ND2p, ND2, m0, n0, lane);
    grid_barrier(ctr, ++barNum * NBLK);

    if (n0 < ND3p)   // f16 feedback -> DIN[t+1]; f32 -> y[b, f, t] = y[(b*F+f)*T + t]
      wave_gemm<0>(dact2, wD2, db2, fb_next, KDp, y + t, (long)F * T, (long)T,
                   ND2p, ND3p, ND3, m0, n0, lane);
    grid_barrier(ctr, ++barNum * NBLK);
  }
}

// ------------------------------ staging kernels -----------------------------
__global__ void pack_w(const float* __restrict__ src, _Float16* __restrict__ dst,
                       int Kreal, int Nreal, int Kp, int Np) {
  size_t idx = (size_t)blockIdx.x * blockDim.x + threadIdx.x;
  const size_t total = (size_t)(Kp >> 5) * 2 * (size_t)Np * 16;
  if (idx >= total) return;
  const int    i = (int)(idx & 15);
  const size_t g = idx >> 4;
  const int    n = (int)(g % (size_t)Np);
  const size_t q = g / (size_t)Np;
  const int    h = (int)(q & 1);
  const int    kt = (int)(q >> 1);
  const int    k = kt * 32 + h * 16 + i;
  float v = (k < Kreal && n < Nreal) ? src[(size_t)k * (size_t)Nreal + n] : 0.f;
  dst[idx] = (_Float16)v;
}

// X0[t*B + b, :] = [ x[b,:,t-1] (0 at t=0) | x[b,:,t] | t/T | zeros ]
__global__ void build_enc_input(const float* __restrict__ x, _Float16* __restrict__ X0) {
  using namespace sw;
  size_t idx = (size_t)blockIdx.x * blockDim.x + threadIdx.x;
  const size_t total = (size_t)ME * (size_t)KEp;
  if (idx >= total) return;
  const int col = (int)(idx % KEp);
  const size_t r = idx / KEp;
  const int b = (int)(r % B);
  const int t = (int)(r / B);
  float v;
  if (col < F)           v = (t == 0) ? 0.f : x[((size_t)b * F + col) * T + (t - 1)];
  else if (col < 2 * F)  v = x[((size_t)b * F + (col - F)) * T + t];
  else if (col == 2 * F) v = (float)t / (float)T;
  else                   v = 0.f;
  X0[idx] = (_Float16)v;
}

// One-shot init of DIN[(T+1)*B, KDp] static parts + reset of barrier counter.
__global__ void init_din(_Float16* __restrict__ DIN, unsigned* __restrict__ ctr) {
  using namespace sw;
  size_t idx = (size_t)blockIdx.x * blockDim.x + threadIdx.x;
  if (idx == 0) *ctr = 0u;
  const size_t total = (size_t)(T + 1) * B * (size_t)KDp;
  if (idx >= total) return;
  const int col = (int)(idx % KDp);
  const int t   = (int)((idx / KDp) / B);
  if (col == H + F)                 DIN[idx] = (_Float16)((float)t / (float)T);
  else if (col > H + F)             DIN[idx] = (_Float16)0.f;   // K padding
  else if (t == 0 && col >= H)      DIN[idx] = (_Float16)0.f;   // zero feedback at t=0
}

// ------------------------------ launcher ------------------------------------
extern "C" void kernel_launch(void* const* d_in, const int* in_sizes, int n_in,
                              void* d_out, int out_size, void* d_ws, size_t ws_size,
                              hipStream_t stream) {
  using namespace sw;
  const float* x   = (const float*)d_in[0];
  const float* ew0 = (const float*)d_in[1];  const float* eb0 = (const float*)d_in[2];
  const float* ew1 = (const float*)d_in[3];  const float* eb1 = (const float*)d_in[4];
  const float* ew2 = (const float*)d_in[5];  const float* eb2 = (const float*)d_in[6];
  const float* dw0 = (const float*)d_in[7];  const float* db0 = (const float*)d_in[8];
  const float* dw1 = (const float*)d_in[9];  const float* db1 = (const float*)d_in[10];
  const float* dw2 = (const float*)d_in[11]; const float* db2 = (const float*)d_in[12];
  float* y = (float*)d_out;

  // --- workspace carve-up (256B aligned) ---
  size_t off = 0;
  auto take = [&](size_t elems) {
    void* p = (char*)d_ws + off;
    off += ((elems * sizeof(_Float16) + 255) / 256) * 256;
    return (_Float16*)p;
  };
  unsigned* ctr = (unsigned*)take(128);                // barrier counter slab
  _Float16* wE0 = take((size_t)KEp  * NE1p);
  _Float16* wE1 = take((size_t)NE1p * NE2p);
  _Float16* wE2 = take((size_t)NE2p * NE3p);
  _Float16* wD0 = take((size_t)KDp  * ND1p);
  _Float16* wD1 = take((size_t)ND1p * ND2p);
  _Float16* wD2 = take((size_t)ND2p * ND3p);
  _Float16* DIN   = take((size_t)(T + 1) * B * KDp);   // per-step decoder inputs
  _Float16* dact1 = take((size_t)B * ND1p);
  _Float16* dact2 = take((size_t)B * ND2p);
  _Float16* X0    = take((size_t)ME * KEp);            // dead after encoder layer 1
  _Float16* A1    = take((size_t)ME * NE1p);
  _Float16* A2    = X0;                                // reuse X0's region

  auto packW = [&](const float* src, _Float16* dst, int Kr, int Nr, int Kp, int Np) {
    size_t total = (size_t)(Kp >> 5) * 2 * (size_t)Np * 16;
    pack_w<<<(unsigned)((total + 255) / 256), 256, 0, stream>>>(src, dst, Kr, Nr, Kp, Np);
  };
  packW(ew0, wE0, KE,  NE1, KEp,  NE1p);
  packW(ew1, wE1, NE1, NE2, NE1p, NE2p);
  packW(ew2, wE2, NE2, NE3, NE2p, NE3p);
  packW(dw0, wD0, KD,  ND1, KDp,  ND1p);
  packW(dw1, wD1, ND1, ND2, ND1p, ND2p);
  packW(dw2, wD2, ND2, ND3, ND2p, ND3p);

  {
    size_t tot = (size_t)ME * KEp;
    build_enc_input<<<(unsigned)((tot + 255) / 256), 256, 0, stream>>>(x, X0);
    size_t dtot = (size_t)(T + 1) * B * KDp;
    init_din<<<(unsigned)((dtot + 255) / 256), 256, 0, stream>>>(DIN, ctr);
  }

  auto gdim = [](int M, int Np) {
    return dim3((unsigned)((Np + 255) / 256), (unsigned)(M / 32));
  };

  // --- encoder: three big WMMA GEMMs over M = T*B = 16384 rows ---
  gemm_f16_wmma<1><<<gdim(ME, NE1p), 256, 0, stream>>>(
      X0, wE0, eb0, A1, NE1p, KEp, NE1p, NE1);
  gemm_f16_wmma<1><<<gdim(ME, NE2p), 256, 0, stream>>>(
      A1, wE1, eb1, A2, NE2p, NE1p, NE2p, NE2);
  // tanh(h) written straight into DIN cols 0..63 (row = t*B+b, ld = KDp)
  gemm_f16_wmma<2><<<gdim(ME, NE3p), 256, 0, stream>>>(
      A2, wE2, eb2, DIN, KDp, NE2p, NE3p, NE3);

  // --- decoder: one persistent kernel, 512 steps, grid barriers between layers
  decoder_persistent<<<dim3(NBLK), 256, 0, stream>>>(
      wD0, db0, wD1, db1, wD2, db2, DIN, dact1, dact2, y, ctr);

  (void)in_sizes; (void)n_in; (void)out_size; (void)ws_size;
}
